// Attention_33346126086636
// MI455X (gfx1250) — matrix operand order repaired and checked by reading, not compiled, not run on hardware
//
#include <hip/hip_runtime.h>

// ---------------- problem constants ----------------
constexpr int B = 2, S = 2048, D = 768, N = 12, H = 64;
constexpr int NH = N * H;                 // 768
constexpr float EPS = 1e-5f;

typedef __bf16 bf16_t;
typedef __attribute__((ext_vector_type(16))) __bf16 v16bf;
typedef __attribute__((ext_vector_type(8)))  __bf16 v8bf;
typedef __attribute__((ext_vector_type(8)))  float  v8f;

static __device__ __forceinline__ v8f zero8() {
  v8f z;
#pragma unroll
  for (int i = 0; i < 8; ++i) z[i] = 0.f;
  return z;
}

static __device__ __forceinline__ v8f wmma_bf16(v16bf a, v16bf b, v8f c) {
  // D = A(16x32 bf16) * B(32x16 bf16) + C(16x16 f32)
  return __builtin_amdgcn_wmma_f32_16x16x32_bf16(
      /*neg_a=*/false, a, /*neg_b=*/false, b,
      /*c_mod=*/(short)0, c, /*reuse_a=*/false, /*reuse_b=*/false);
}

// A fragment (16-bit, 16x32): lane L holds row M=L&15.
// elements 0..7  = K = k0 + 8*(L>>4) + e
// elements 8..15 = K = k0 + 16 + 8*(L>>4) + e
static __device__ __forceinline__ v16bf load_afrag(const bf16_t* rowbase, int ld,
                                                   int k0, int lane) {
  const int m = lane & 15, h = lane >> 4;
  const bf16_t* p = rowbase + (size_t)m * ld + k0 + 8 * h;
  v8bf lo = *(const v8bf*)p;
  v8bf hi = *(const v8bf*)(p + 16);
  v16bf a;
#pragma unroll
  for (int e = 0; e < 8; ++e) { a[e] = lo[e]; a[e + 8] = hi[e]; }
  return a;
}

// B fragment (16-bit, 32x16) from K-major (transposed) storage:
// lane L holds column n=L&15; elements 0..7: K = k0 + 8*(L>>4) + e,
// elements 8..15: K = k0 + 16 + 8*(L>>4) + e.
static __device__ __forceinline__ v16bf load_bfrag(const bf16_t* colbase, int ld,
                                                   int k0, int lane) {
  const int nn = lane & 15, h = lane >> 4;
  const bf16_t* p = colbase + (size_t)nn * ld + k0 + 8 * h;
  v8bf lo = *(const v8bf*)p;
  v8bf hi = *(const v8bf*)(p + 16);
  v16bf r;
#pragma unroll
  for (int e = 0; e < 8; ++e) { r[e] = lo[e]; r[e + 8] = hi[e]; }
  return r;
}

static __device__ __forceinline__ float wave_red_add(float v) {
#pragma unroll
  for (int off = 16; off > 0; off >>= 1) v += __shfl_xor(v, off, 32);
  return v;
}

// ---------------- kernel 1: fp32 -> bf16 activation pack ----------------
__global__ void __launch_bounds__(256)
pack_x_kernel(const float* __restrict__ xq, const float* __restrict__ xkv,
              bf16_t* __restrict__ xqb, bf16_t* __restrict__ xkvb) {
  size_t i = (size_t)blockIdx.x * blockDim.x + threadIdx.x;
  if (i < (size_t)B * S * D) {
    xqb[i]  = (bf16_t)xq[i];
    xkvb[i] = (bf16_t)xkv[i];
  }
}

// ---------------- kernel 2: weight transpose-pack ----------------
// W_Q/K/V [N][D][H] -> [N][H][D] bf16 (B-frag friendly: contiguous over d)
// W_O     [N][H][D] -> [D][N*H]  bf16 (contiguous over n*h)
__global__ void __launch_bounds__(256)
pack_w_kernel(const float* __restrict__ Wq, const float* __restrict__ Wk,
              const float* __restrict__ Wv, const float* __restrict__ Wo,
              bf16_t* __restrict__ WqT, bf16_t* __restrict__ WkT,
              bf16_t* __restrict__ WvT, bf16_t* __restrict__ WoT) {
  int i = blockIdx.x * blockDim.x + threadIdx.x;
  if (i >= N * D * H) return;
  {  // i interpreted as [n][d][h]
    int hh = i % H, dd = (i / H) % D, nn = i / (H * D);
    size_t o = ((size_t)nn * H + hh) * D + dd;
    WqT[o] = (bf16_t)Wq[i];
    WkT[o] = (bf16_t)Wk[i];
    WvT[o] = (bf16_t)Wv[i];
  }
  {  // i interpreted as [n][h][d]
    int dd = i % D, hh = (i / D) % H, nn = i / (D * H);
    WoT[(size_t)dd * NH + nn * H + hh] = (bf16_t)Wo[i];
  }
}

// ---------------- kernel 3a: Q projection (2x2 blocking, 4 accums) ---------
__global__ void __launch_bounds__(128)
q_proj_kernel(const bf16_t* __restrict__ xq, const bf16_t* __restrict__ WqT,
              float* __restrict__ Qraw) {
  const int lane = threadIdx.x & 31;
  const int w = blockIdx.x * (blockDim.x >> 5) + (threadIdx.x >> 5);
  const int ht2 = w & 1;              // H/32 = 2
  const int st2 = (w >> 1) & 63;      // S/32 = 64
  const int r2 = w >> 7;
  const int n = r2 % N, b = r2 / N;
  const int sBase = st2 * 32, hBase = ht2 * 32;

  const bf16_t* x0 = xq + ((size_t)b * S + sBase) * D;
  const bf16_t* x1 = x0 + (size_t)16 * D;
  const bf16_t* w0 = WqT + ((size_t)n * H + hBase) * D;
  const bf16_t* w1 = w0 + (size_t)16 * D;

  v8f c00 = zero8(), c01 = zero8(), c10 = zero8(), c11 = zero8();
  for (int k0 = 0; k0 < D; k0 += 32) {
    v16bf a0 = load_afrag(x0, D, k0, lane);
    v16bf a1 = load_afrag(x1, D, k0, lane);
    v16bf b0 = load_bfrag(w0, D, k0, lane);
    v16bf b1 = load_bfrag(w1, D, k0, lane);
    c00 = wmma_bf16(a0, b0, c00);  c01 = wmma_bf16(a0, b1, c01);
    c10 = wmma_bf16(a1, b0, c10);  c11 = wmma_bf16(a1, b1, c11);
  }
  const int col = lane & 15, half = lane >> 4;
  const size_t base = ((size_t)(b * N + n)) * S * H;
#pragma unroll
  for (int r = 0; r < 8; ++r) {
    int row0 = sBase + r + 8 * half;
    size_t o0 = base + (size_t)row0 * H + hBase + col;
    size_t o1 = o0 + (size_t)16 * H;
    Qraw[o0] = c00[r];  Qraw[o0 + 16] = c01[r];
    Qraw[o1] = c10[r];  Qraw[o1 + 16] = c11[r];
  }
}

// ---------------- kernel 3b: KV projection (2x2 blocking, 8 accums) --------
__global__ void __launch_bounds__(128)
kv_proj_kernel(const bf16_t* __restrict__ xkv, const bf16_t* __restrict__ WkT,
               const bf16_t* __restrict__ WvT, float* __restrict__ Kraw,
               float* __restrict__ Vraw) {
  const int lane = threadIdx.x & 31;
  const int w = blockIdx.x * (blockDim.x >> 5) + (threadIdx.x >> 5);
  const int ht2 = w & 1;              // H/32 = 2
  const int st2 = (w >> 1) & 63;      // S/32 = 64
  const int r2 = w >> 7;
  const int n = r2 % N, b = r2 / N;
  const int sBase = st2 * 32, hBase = ht2 * 32;

  const bf16_t* x0 = xkv + ((size_t)b * S + sBase) * D;
  const bf16_t* x1 = x0 + (size_t)16 * D;
  const bf16_t* wk0 = WkT + ((size_t)n * H + hBase) * D;
  const bf16_t* wk1 = wk0 + (size_t)16 * D;
  const bf16_t* wv0 = WvT + ((size_t)n * H + hBase) * D;
  const bf16_t* wv1 = wv0 + (size_t)16 * D;

  v8f ck00 = zero8(), ck01 = zero8(), ck10 = zero8(), ck11 = zero8();
  v8f cv00 = zero8(), cv01 = zero8(), cv10 = zero8(), cv11 = zero8();
  for (int k0 = 0; k0 < D; k0 += 32) {
    v16bf a0 = load_afrag(x0, D, k0, lane);
    v16bf a1 = load_afrag(x1, D, k0, lane);
    v16bf bk0 = load_bfrag(wk0, D, k0, lane);
    v16bf bk1 = load_bfrag(wk1, D, k0, lane);
    ck00 = wmma_bf16(a0, bk0, ck00);  ck01 = wmma_bf16(a0, bk1, ck01);
    ck10 = wmma_bf16(a1, bk0, ck10);  ck11 = wmma_bf16(a1, bk1, ck11);
    v16bf bv0 = load_bfrag(wv0, D, k0, lane);
    v16bf bv1 = load_bfrag(wv1, D, k0, lane);
    cv00 = wmma_bf16(a0, bv0, cv00);  cv01 = wmma_bf16(a0, bv1, cv01);
    cv10 = wmma_bf16(a1, bv0, cv10);  cv11 = wmma_bf16(a1, bv1, cv11);
  }
  const int col = lane & 15, half = lane >> 4;
  const size_t base = ((size_t)(b * N + n)) * S * H;
#pragma unroll
  for (int r = 0; r < 8; ++r) {
    int row0 = sBase + r + 8 * half;
    size_t o0 = base + (size_t)row0 * H + hBase + col;
    size_t o1 = o0 + (size_t)16 * H;
    Kraw[o0] = ck00[r];  Kraw[o0 + 16] = ck01[r];
    Kraw[o1] = ck10[r];  Kraw[o1 + 16] = ck11[r];
    Vraw[o0] = cv00[r];  Vraw[o0 + 16] = cv01[r];
    Vraw[o1] = cv10[r];  Vraw[o1 + 16] = cv11[r];
  }
}

// ---------------- kernel 4: per-head LayerNorm (Q,K) + V transpose ----------
// one wave per (b,n,s) row of 64; lane holds elements [lane] and [lane+32].
__global__ void __launch_bounds__(128)
ln_pack_kernel(const float* __restrict__ Qraw, const float* __restrict__ Kraw,
               const float* __restrict__ Vraw,
               const float* __restrict__ g1, const float* __restrict__ b1,
               const float* __restrict__ g2, const float* __restrict__ b2,
               bf16_t* __restrict__ Qbf, bf16_t* __restrict__ Kbf,
               bf16_t* __restrict__ Vt) {
  const int lane = threadIdx.x & 31;
  const int row = blockIdx.x * (blockDim.x >> 5) + (threadIdx.x >> 5);  // b*n*s
  const size_t base = (size_t)row * H;

  {  // Q with ln1
    float q0 = Qraw[base + lane], q1 = Qraw[base + lane + 32];
    float mu = wave_red_add(q0 + q1) * (1.f / 64.f);
    float d0 = q0 - mu, d1 = q1 - mu;
    float var = wave_red_add(d0 * d0 + d1 * d1) * (1.f / 64.f);
    float rs = rsqrtf(var + EPS);
    Qbf[base + lane]      = (bf16_t)(d0 * rs * g1[lane]      + b1[lane]);
    Qbf[base + lane + 32] = (bf16_t)(d1 * rs * g1[lane + 32] + b1[lane + 32]);
  }
  {  // K with ln2
    float k0 = Kraw[base + lane], k1 = Kraw[base + lane + 32];
    float mu = wave_red_add(k0 + k1) * (1.f / 64.f);
    float d0 = k0 - mu, d1 = k1 - mu;
    float var = wave_red_add(d0 * d0 + d1 * d1) * (1.f / 64.f);
    float rs = rsqrtf(var + EPS);
    Kbf[base + lane]      = (bf16_t)(d0 * rs * g2[lane]      + b2[lane]);
    Kbf[base + lane + 32] = (bf16_t)(d1 * rs * g2[lane + 32] + b2[lane + 32]);
  }
  {  // V -> Vt[b][n][h][s] bf16
    int s_idx = row % S, bn = row / S;
    Vt[((size_t)bn * H + lane) * S + s_idx]      = (bf16_t)Vraw[base + lane];
    Vt[((size_t)bn * H + lane + 32) * S + s_idx] = (bf16_t)Vraw[base + lane + 32];
  }
}

// ---------------- kernel 5: causal flash attention (32 queries / wave) -----
// Scores computed transposed (key x query) so the softmaxed D-fragments ARE
// the A-fragments for P·V. K-fragments reused across both query tiles,
// V-fragments reused across both P tiles: 8 frag loads per 16 WMMAs.
// Softmax is serialized per query tile to bound live VGPRs.
__global__ void __launch_bounds__(128)
attn_kernel(const bf16_t* __restrict__ Qbf, const bf16_t* __restrict__ Kbf,
            const bf16_t* __restrict__ Vt, bf16_t* __restrict__ Z) {
  const int lane = threadIdx.x & 31;
  const int w = blockIdx.x * (blockDim.x >> 5) + (threadIdx.x >> 5);
  const int qt = w & 63;              // S/32
  const int tmp = w >> 6;
  const int n = tmp % N, b = tmp / N;
  const int qBase = qt * 32;
  const int col = lane & 15, half = lane >> 4;

  const bf16_t* Qb = Qbf + (size_t)(b * N + n) * S * H;
  const bf16_t* Kb = Kbf + (size_t)(b * N + n) * S * H;
  const bf16_t* Vb = Vt  + (size_t)(b * N + n) * H * S;

  // Q as B-fragments for both 16-query tiles (loaded once).
  v16bf bqA0 = load_bfrag(Qb + (size_t)qBase * H,        H, 0,  lane);
  v16bf bqA1 = load_bfrag(Qb + (size_t)qBase * H,        H, 32, lane);
  v16bf bqB0 = load_bfrag(Qb + (size_t)(qBase + 16) * H, H, 0,  lane);
  v16bf bqB1 = load_bfrag(Qb + (size_t)(qBase + 16) * H, H, 32, lane);

  const int qIdxA = qBase + col, qIdxB = qBase + 16 + col;
  float mA = -1e30f, mB = -1e30f, lA = 0.f, lB = 0.f;
  v8f oA0 = zero8(), oA1 = zero8(), oA2 = zero8(), oA3 = zero8();
  v8f oB0 = zero8(), oB1 = zero8(), oB2 = zero8(), oB3 = zero8();

  for (int kb = 0; kb < qBase + 32; kb += 32) {
    if (kb + 32 < S)  // pull next K block toward L0/L2
      __builtin_prefetch(Kb + (size_t)(kb + 32 + lane) * H, 0, 1);

    // K A-fragments for the two 16-key tiles (shared by both query tiles).
    v16bf ak0_lo = load_afrag(Kb + (size_t)kb * H,        H, 0,  lane);
    v16bf ak0_hi = load_afrag(Kb + (size_t)kb * H,        H, 32, lane);
    v16bf ak1_lo = load_afrag(Kb + (size_t)(kb + 16) * H, H, 0,  lane);
    v16bf ak1_hi = load_afrag(Kb + (size_t)(kb + 16) * H, H, 32, lane);

    v16bf apA, apB;

    {  // ---- query tile A: scores, mask, online softmax -> apA ----
      v8f s0, s1;
      { v8f z = zero8(); z = wmma_bf16(ak0_lo, bqA0, z); s0 = wmma_bf16(ak0_hi, bqA1, z); }
      { v8f z = zero8(); z = wmma_bf16(ak1_lo, bqA0, z); s1 = wmma_bf16(ak1_hi, bqA1, z); }
      float mt = -1e30f;
#pragma unroll
      for (int r = 0; r < 8; ++r) {
        int key0 = kb + r + 8 * half;
        if (key0 > qIdxA)      s0[r] = -1e30f;
        if (key0 + 16 > qIdxA) s1[r] = -1e30f;
        mt = fmaxf(mt, fmaxf(s0[r], s1[r]));
      }
      mt = fmaxf(mt, __shfl_xor(mt, 16, 32));
      float mn = fmaxf(mA, mt);
      float al = __expf(mA - mn);
      float ps = 0.f;
#pragma unroll
      for (int r = 0; r < 8; ++r) {
        float p0 = __expf(s0[r] - mn), p1 = __expf(s1[r] - mn);
        ps += p0 + p1;
        apA[r] = (bf16_t)p0;  apA[r + 8] = (bf16_t)p1;
      }
      ps += __shfl_xor(ps, 16, 32);
      lA = lA * al + ps;  mA = mn;
#pragma unroll
      for (int r = 0; r < 8; ++r) {
        float ar = __shfl(al, r + 8 * half, 32);
        oA0[r] *= ar; oA1[r] *= ar; oA2[r] *= ar; oA3[r] *= ar;
      }
    }

    {  // ---- query tile B: scores, mask, online softmax -> apB ----
      v8f s0, s1;
      { v8f z = zero8(); z = wmma_bf16(ak0_lo, bqB0, z); s0 = wmma_bf16(ak0_hi, bqB1, z); }
      { v8f z = zero8(); z = wmma_bf16(ak1_lo, bqB0, z); s1 = wmma_bf16(ak1_hi, bqB1, z); }
      float mt = -1e30f;
#pragma unroll
      for (int r = 0; r < 8; ++r) {
        int key0 = kb + r + 8 * half;
        if (key0 > qIdxB)      s0[r] = -1e30f;
        if (key0 + 16 > qIdxB) s1[r] = -1e30f;
        mt = fmaxf(mt, fmaxf(s0[r], s1[r]));
      }
      mt = fmaxf(mt, __shfl_xor(mt, 16, 32));
      float mn = fmaxf(mB, mt);
      float al = __expf(mB - mn);
      float ps = 0.f;
#pragma unroll
      for (int r = 0; r < 8; ++r) {
        float p0 = __expf(s0[r] - mn), p1 = __expf(s1[r] - mn);
        ps += p0 + p1;
        apB[r] = (bf16_t)p0;  apB[r + 8] = (bf16_t)p1;
      }
      ps += __shfl_xor(ps, 16, 32);
      lB = lB * al + ps;  mB = mn;
#pragma unroll
      for (int r = 0; r < 8; ++r) {
        float ar = __shfl(al, r + 8 * half, 32);
        oB0[r] *= ar; oB1[r] *= ar; oB2[r] *= ar; oB3[r] *= ar;
      }
    }

    // V B-fragments loaded once, used by both query tiles.
    {
      v16bf bv = load_bfrag(Vb + (size_t)0 * S, S, kb, lane);
      oA0 = wmma_bf16(apA, bv, oA0);  oB0 = wmma_bf16(apB, bv, oB0);
    }
    {
      v16bf bv = load_bfrag(Vb + (size_t)16 * S, S, kb, lane);
      oA1 = wmma_bf16(apA, bv, oA1);  oB1 = wmma_bf16(apB, bv, oB1);
    }
    {
      v16bf bv = load_bfrag(Vb + (size_t)32 * S, S, kb, lane);
      oA2 = wmma_bf16(apA, bv, oA2);  oB2 = wmma_bf16(apB, bv, oB2);
    }
    {
      v16bf bv = load_bfrag(Vb + (size_t)48 * S, S, kb, lane);
      oA3 = wmma_bf16(apA, bv, oA3);  oB3 = wmma_bf16(apB, bv, oB3);
    }
  }

  // epilogue: divide by l (per query = per register) and store Z[b][s][n*H+h]
#pragma unroll
  for (int r = 0; r < 8; ++r) {
    float invA = 1.0f / __shfl(lA, r + 8 * half, 32);
    float invB = 1.0f / __shfl(lB, r + 8 * half, 32);
    int srowA = qBase + r + 8 * half;
    int srowB = srowA + 16;
    size_t offA = ((size_t)b * S + srowA) * NH + (size_t)n * H;
    size_t offB = ((size_t)b * S + srowB) * NH + (size_t)n * H;
    Z[offA +  0 + col] = (bf16_t)(oA0[r] * invA);
    Z[offA + 16 + col] = (bf16_t)(oA1[r] * invA);
    Z[offA + 32 + col] = (bf16_t)(oA2[r] * invA);
    Z[offA + 48 + col] = (bf16_t)(oA3[r] * invA);
    Z[offB +  0 + col] = (bf16_t)(oB0[r] * invB);
    Z[offB + 16 + col] = (bf16_t)(oB1[r] * invB);
    Z[offB + 32 + col] = (bf16_t)(oB2[r] * invB);
    Z[offB + 48 + col] = (bf16_t)(oB3[r] * invB);
  }
}

// ---------------- kernel 6: output projection (WMMA GEMM, 2x2 blocking) ----
// out[b][s][d] = Z[b][s][:] (768) @ WoT^T ; 4 loads per 4 WMMAs per K-step.
__global__ void __launch_bounds__(128)
out_proj_kernel(const bf16_t* __restrict__ Z, const bf16_t* __restrict__ WoT,
                float* __restrict__ out) {
  const int lane = threadIdx.x & 31;
  const int w = blockIdx.x * (blockDim.x >> 5) + (threadIdx.x >> 5);
  const int dt2 = w % (D / 32);       // 24
  const int rest = w / (D / 32);
  const int st2 = rest % (S / 32);    // 64
  const int b = rest / (S / 32);
  const int sBase = st2 * 32, dBase = dt2 * 32;

  const bf16_t* Zr0 = Z + ((size_t)b * S + sBase) * NH;
  const bf16_t* Zr1 = Zr0 + (size_t)16 * NH;
  const bf16_t* Wr0 = WoT + (size_t)dBase * NH;
  const bf16_t* Wr1 = Wr0 + (size_t)16 * NH;

  v8f c00 = zero8(), c01 = zero8(), c10 = zero8(), c11 = zero8();
  for (int k0 = 0; k0 < NH; k0 += 32) {
    v16bf a0 = load_afrag(Zr0, NH, k0, lane);
    v16bf a1 = load_afrag(Zr1, NH, k0, lane);
    v16bf b0 = load_bfrag(Wr0, NH, k0, lane);
    v16bf b1 = load_bfrag(Wr1, NH, k0, lane);
    c00 = wmma_bf16(a0, b0, c00);  c01 = wmma_bf16(a0, b1, c01);
    c10 = wmma_bf16(a1, b0, c10);  c11 = wmma_bf16(a1, b1, c11);
  }
  const int col = lane & 15, half = lane >> 4;
#pragma unroll
  for (int r = 0; r < 8; ++r) {
    int srow0 = sBase + r + 8 * half;
    int srow1 = srow0 + 16;
    size_t o0 = ((size_t)b * S + srow0) * D + dBase + col;
    size_t o1 = ((size_t)b * S + srow1) * D + dBase + col;
    out[o0] = c00[r];  out[o0 + 16] = c01[r];
    out[o1] = c10[r];  out[o1 + 16] = c11[r];
  }
}

// ---------------- host orchestration ----------------
extern "C" void kernel_launch(void* const* d_in, const int* in_sizes, int n_in,
                              void* d_out, int out_size, void* d_ws, size_t ws_size,
                              hipStream_t stream) {
  const float* x_q  = (const float*)d_in[0];
  const float* x_kv = (const float*)d_in[1];
  // d_in[2] = mask (causal, regenerated analytically)
  const float* W_Q = (const float*)d_in[3];
  const float* W_K = (const float*)d_in[4];
  const float* W_V = (const float*)d_in[5];
  const float* W_O = (const float*)d_in[6];
  const float* g1 = (const float*)d_in[7];
  const float* b1 = (const float*)d_in[8];
  const float* g2 = (const float*)d_in[9];
  const float* b2 = (const float*)d_in[10];
  float* out = (float*)d_out;

  char* ws = (char*)d_ws;
  size_t off = 0;
  auto carve = [&](size_t bytes) -> char* {
    char* p = ws + off;
    off = (off + bytes + 255) & ~(size_t)255;
    return p;
  };
  const size_t nX  = (size_t)B * S * D;       // 3,145,728
  const size_t nW  = (size_t)N * D * H;       //   589,824
  const size_t nBH = (size_t)B * N * S * H;   // 3,145,728

  bf16_t* xqb  = (bf16_t*)carve(nX * 2);
  bf16_t* xkvb = (bf16_t*)carve(nX * 2);
  bf16_t* WqT  = (bf16_t*)carve(nW * 2);
  bf16_t* WkT  = (bf16_t*)carve(nW * 2);
  bf16_t* WvT  = (bf16_t*)carve(nW * 2);
  bf16_t* WoT  = (bf16_t*)carve(nW * 2);
  float*  Qraw = (float*)carve(nBH * 4);
  float*  Kraw = (float*)carve(nBH * 4);
  float*  Vraw = (float*)carve(nBH * 4);
  bf16_t* Qbf  = (bf16_t*)carve(nBH * 2);
  bf16_t* Kbf  = (bf16_t*)carve(nBH * 2);
  bf16_t* Vt   = (bf16_t*)carve(nBH * 2);
  bf16_t* Zbuf = (bf16_t*)carve((size_t)B * S * NH * 2);
  (void)ws_size; (void)in_sizes; (void)n_in; (void)out_size;

  // 1) activation + weight packing
  pack_x_kernel<<<(int)((nX + 255) / 256), 256, 0, stream>>>(x_q, x_kv, xqb, xkvb);
  pack_w_kernel<<<(int)((nW + 255) / 256), 256, 0, stream>>>(W_Q, W_K, W_V, W_O,
                                                             WqT, WkT, WvT, WoT);
  // 2) Q / KV projections: B*N*(S/32)*(H/32) waves each, 4 waves/block
  {
    int waves = B * N * (S / 32) * (H / 32);   // 3072
    q_proj_kernel<<<waves / 4, 128, 0, stream>>>(xqb, WqT, Qraw);
    kv_proj_kernel<<<waves / 4, 128, 0, stream>>>(xkvb, WkT, WvT, Kraw, Vraw);
  }
  // 3) LayerNorm + pack: one wave per (b,n,s) row
  {
    int rows = B * N * S;                       // 49152
    ln_pack_kernel<<<rows / 4, 128, 0, stream>>>(Qraw, Kraw, Vraw, g1, b1, g2, b2,
                                                 Qbf, Kbf, Vt);
  }
  // 4) flash attention: B*N*(S/32) waves
  {
    int waves = B * N * (S / 32);               // 1536
    attn_kernel<<<waves / 4, 128, 0, stream>>>(Qbf, Kbf, Vt, Zbuf);
  }
  // 5) output projection: B*(S/32)*(D/32) waves
  {
    int waves = B * (S / 32) * (D / 32);        // 3072
    out_proj_kernel<<<waves / 4, 128, 0, stream>>>(Zbuf, WoT, out);
  }
}
